// SimpleGNNBinary_34797825032947
// MI455X (gfx1250) — compile-verified, matrix operand order checked by reading, not run on hardware
//
#include <hip/hip_runtime.h>
#include <hip/hip_bf16.h>
#include <math.h>

// ---------------------------------------------------------------------------
// MI455X (gfx1250) GNN forward.  All dense per-edge / per-node MLPs run on
// v_wmma_f32_16x16x32_bf16 (wave32, one wave = one 16-row tile, K=128 -> 4
// WMMA per 16x16 C tile).  fp32 accumulation, fp32 segment-sum + residual.
// Weights are pre-swizzled into B-fragment order, block-staged into LDS
// (ds_load feeds the WMMA pipe instead of global loads), and tiles are
// processed persistently (grid-stride) to amortize the staging.
// ---------------------------------------------------------------------------

typedef __attribute__((ext_vector_type(16))) __bf16 v16bf;
typedef __attribute__((ext_vector_type(8)))  __bf16 v8bf;
typedef __attribute__((ext_vector_type(8)))  float  v8f;

#define HD  64   // hidden size
#define NF  18   // node feature size
#define WPB 4    // waves per block (128 threads)

// packed fragment counts (ushorts)
#define PM1 (4 * 4 * 512)  // 128x64
#define PM2 (2 * 4 * 512)  // 64x64
#define PA1 (4 * 2 * 512)  // 128x32

__device__ __forceinline__ unsigned short f2bf(float f) {
  unsigned u = __float_as_uint(f);
  u += 0x7FFFu + ((u >> 16) & 1u);           // round-to-nearest-even
  return (unsigned short)(u >> 16);
}

// A-fragment (16x32 bf16, ISA 7.12.2): lane m = lane&15, per-lane K indices
// are {b..b+7, b+16..b+23} with b = kbase + (lane>=16 ? 8 : 0).
__device__ __forceinline__ v16bf load_A_bf16(const unsigned short* rowptr, int kbase, int lane) {
  int b = kbase + ((lane >> 4) << 3);
  v8bf lo = *(const v8bf*)(rowptr + b);
  v8bf hi = *(const v8bf*)(rowptr + b + 16);
  v16bf a;
#pragma unroll
  for (int i = 0; i < 8; ++i) { a[i] = lo[i]; a[8 + i] = hi[i]; }
  return a;
}

// Same A layout but sourced from fp32 (aggregated messages), scaled, cvt bf16.
__device__ __forceinline__ v16bf load_A_f32_scaled(const float* rowptr, int kbase, int lane, float scale) {
  int b = kbase + ((lane >> 4) << 3);
  unsigned short t[16];
#pragma unroll
  for (int i = 0; i < 8; ++i) {
    t[i]     = f2bf(rowptr[b + i] * scale);
    t[8 + i] = f2bf(rowptr[b + 16 + i] * scale);
  }
  v16bf a; __builtin_memcpy(&a, t, sizeof(a));
  return a;
}

// B fragments: pre-packed so each lane's 16 bf16 values are contiguous (32B).
// Served out of LDS (ds_load_b128 x2) after block staging.
__device__ __forceinline__ v16bf load_B_frag(const unsigned short* packed, int frag, int lane) {
  return *(const v16bf*)(packed + (((long)frag << 5) + lane) * 16);
}

__device__ __forceinline__ v8f wmma_bf16(v16bf a, v16bf b, v8f c) {
  return __builtin_amdgcn_wmma_f32_16x16x32_bf16(false, a, false, b, (short)0, c, false, false);
}

// cooperative global->LDS copy, 16B per thread per step (count % 1024 == 0)
__device__ __forceinline__ void stage_lds(unsigned short* dst, const unsigned short* src,
                                          int count_ushort, int tid, int nthr) {
  const uint4* s = (const uint4*)src;
  uint4* d = (uint4*)dst;
  int n = count_ushort >> 3;               // uint4 = 8 ushorts
  for (int i = tid; i < n; i += nthr) d[i] = s[i];
}

// ---------------------------------------------------------------------------
// Weight pre-swizzle: fp32 row-major [K][N] -> bf16 B-fragments.
// ---------------------------------------------------------------------------
__global__ void pack_weights_kernel(const float* __restrict__ src,
                                    unsigned short* __restrict__ dst,
                                    int K, int N) {
  int total = (K >> 5) * (N >> 4) * 512;
  for (int t = blockIdx.x * blockDim.x + threadIdx.x; t < total;
       t += gridDim.x * blockDim.x) {
    int i = t & 15, lane = (t >> 4) & 31, frag = t >> 9;
    int nfr = N >> 4;
    int nb = frag % nfr, kb = frag / nfr;
    int kl = (i & 7) + ((i >> 3) << 4) + ((lane >> 4) << 3);
    int k = (kb << 5) + kl;
    int n = (nb << 4) + (lane & 15);
    dst[t] = f2bf(src[k * N + n]);
  }
}

__global__ void zero_f32_kernel(float* __restrict__ p, long n) {
  long t = (long)blockIdx.x * blockDim.x + threadIdx.x;
  if (t < n) p[t] = 0.f;
}
__global__ void zero_u32_kernel(unsigned* __restrict__ p, int n) {
  int t = blockIdx.x * blockDim.x + threadIdx.x;
  if (t < n) p[t] = 0u;
}

// x = relu(nf @ embed_W + b); write fp32 + bf16 mirror.
__global__ void embed_kernel(const float* __restrict__ nf,
                             const float* __restrict__ W,
                             const float* __restrict__ b,
                             float* __restrict__ xf,
                             unsigned short* __restrict__ xb, int N) {
  long t = (long)blockIdx.x * blockDim.x + threadIdx.x;
  long total = (long)N * HD;
  if (t >= total) return;
  int h = (int)(t & (HD - 1));
  long node = t >> 6;
  float s = b[h];
  const float* f = nf + node * NF;
#pragma unroll
  for (int k = 0; k < NF; ++k) s += f[k] * W[k * HD + h];
  s = fmaxf(s, 0.f);
  xf[t] = s;
  xb[t] = f2bf(s);
}

// ---------------------------------------------------------------------------
// Edge kernel (persistent): one wave = one 16-edge tile per loop iteration.
//   cat = [x[row] , x[col]]            (4 bf16 A-fragments, gathered)
//   h1  = relu(cat @ msg_W1 + b1)      } fused K-loop, 6 indep WMMA / K-step
//   ha  = relu(cat @ att_W1 + b1)      }
//   fw  = ew * sigmoid(ha @ att_W2+b2) VALU (lanes 0..15)
//   msg = (h1 @ msg_W2 + b2) * fw      atomicAdd scatter to agg/norm
// ---------------------------------------------------------------------------
__global__ void __launch_bounds__(128)
edge_kernel(const unsigned short* __restrict__ xb,
            const int* __restrict__ row, const int* __restrict__ col,
            const float* __restrict__ ew,
            const unsigned short* __restrict__ pw_msg1, const float* __restrict__ msg_b1,
            const unsigned short* __restrict__ pw_msg2, const float* __restrict__ msg_b2,
            const unsigned short* __restrict__ pw_att1, const float* __restrict__ att_b1,
            const float* __restrict__ att_W2, const float* __restrict__ att_b2,
            float* __restrict__ agg, float* __restrict__ nrm,
            int E, int ntiles) {
  __shared__ unsigned short w_msg1[PM1];   // 16 KB
  __shared__ unsigned short w_msg2[PM2];   //  8 KB
  __shared__ unsigned short w_att1[PA1];   //  8 KB
  __shared__ unsigned short s_h1[WPB][16 * HD];
  __shared__ float s_ha[WPB][16 * 32];
  __shared__ float s_fw[WPB][16];
  __shared__ int   s_col[WPB][16];

  const int tid = threadIdx.x;
  const int wave = tid >> 5;
  const int lane = tid & 31;

  // warm L2 + stage all weight fragments into LDS (before any divergence)
  __builtin_prefetch(pw_msg1, 0, 2);
  __builtin_prefetch(pw_msg2, 0, 2);
  __builtin_prefetch(pw_att1, 0, 2);
  stage_lds(w_msg1, pw_msg1, PM1, tid, 128);
  stage_lds(w_msg2, pw_msg2, PM2, tid, 128);
  stage_lds(w_att1, pw_att1, PA1, tid, 128);
  __syncthreads();

  const int m = lane & 15;
  const int mrb = (lane >> 4) << 3;
  const int nl = lane & 15;

  for (int tile = blockIdx.x * WPB + wave; tile < ntiles;
       tile += gridDim.x * WPB) {
    long e = (long)tile * 16 + m;
    const bool valid = e < (long)E;
    if (!valid) e = (long)E - 1;         // clamp: keep EXEC all-1s for WMMA
    const int ri = row[e], ci = col[e];
    const unsigned short* pr = xb + (long)ri * HD;
    const unsigned short* pc = xb + (long)ci * HD;

    v16bf a[4];
    a[0] = load_A_bf16(pr, 0, lane);
    a[1] = load_A_bf16(pr, 32, lane);
    a[2] = load_A_bf16(pc, 0, lane);
    a[3] = load_A_bf16(pc, 32, lane);

    // fused GEMM1: msg hidden [16,128]x[128,64] + att hidden [16,128]x[128,32]
    v8f accM0 = {}, accM1 = {}, accM2 = {}, accM3 = {};
    v8f accA0 = {}, accA1 = {};
#pragma unroll
    for (int kb = 0; kb < 4; ++kb) {
      v16bf bm0 = load_B_frag(w_msg1, kb * 4 + 0, lane);
      v16bf bm1 = load_B_frag(w_msg1, kb * 4 + 1, lane);
      v16bf bm2 = load_B_frag(w_msg1, kb * 4 + 2, lane);
      v16bf bm3 = load_B_frag(w_msg1, kb * 4 + 3, lane);
      v16bf ba0 = load_B_frag(w_att1, kb * 2 + 0, lane);
      v16bf ba1 = load_B_frag(w_att1, kb * 2 + 1, lane);
      accM0 = wmma_bf16(a[kb], bm0, accM0);
      accM1 = wmma_bf16(a[kb], bm1, accM1);
      accM2 = wmma_bf16(a[kb], bm2, accM2);
      accM3 = wmma_bf16(a[kb], bm3, accM3);
      accA0 = wmma_bf16(a[kb], ba0, accA0);
      accA1 = wmma_bf16(a[kb], ba1, accA1);
    }

    // epilogues -> per-wave LDS staging
    {
      v8f accM[4] = {accM0, accM1, accM2, accM3};
#pragma unroll
      for (int nb = 0; nb < 4; ++nb) {
        int n = nb * 16 + nl;
        float bias = msg_b1[n];
#pragma unroll
        for (int r = 0; r < 8; ++r)
          s_h1[wave][(r + mrb) * HD + n] = f2bf(fmaxf(accM[nb][r] + bias, 0.f));
      }
      v8f accA[2] = {accA0, accA1};
#pragma unroll
      for (int nb = 0; nb < 2; ++nb) {
        int n = nb * 16 + nl;
        float bias = att_b1[n];
#pragma unroll
        for (int r = 0; r < 8; ++r)
          s_ha[wave][(r + mrb) * 32 + n] = fmaxf(accA[nb][r] + bias, 0.f);
      }
    }

    // same-wave LDS write->read (cross-lane): DS pipe is in-order; fence the
    // compiler and drain DScnt explicitly.
    asm volatile("s_wait_dscnt 0" ::: "memory");

    // attention head + edge-weight fusion (lanes 0..15, one edge each)
    if (lane < 16) {
      float s = att_b2[0];
#pragma unroll
      for (int k = 0; k < 32; ++k) s += s_ha[wave][lane * 32 + k] * att_W2[k];
      float atv = 1.f / (1.f + __expf(-s));
      float fw = valid ? ew[e] * atv : 0.f;
      s_fw[wave][lane] = fw;
      s_col[wave][lane] = ci;
      atomicAdd(&nrm[ci], fw);
    }
    asm volatile("s_wait_dscnt 0" ::: "memory");

    // GEMM2: messages [16,64]x[64,64]; 4 independent tiles per K-step
    const unsigned short* hrow = &s_h1[wave][m * HD];
    v16bf h[2];
    h[0] = load_A_bf16(hrow, 0, lane);
    h[1] = load_A_bf16(hrow, 32, lane);
    v8f acc0 = {}, acc1 = {}, acc2 = {}, acc3 = {};
#pragma unroll
    for (int kb = 0; kb < 2; ++kb) {
      v16bf b0 = load_B_frag(w_msg2, kb * 4 + 0, lane);
      v16bf b1 = load_B_frag(w_msg2, kb * 4 + 1, lane);
      v16bf b2 = load_B_frag(w_msg2, kb * 4 + 2, lane);
      v16bf b3 = load_B_frag(w_msg2, kb * 4 + 3, lane);
      acc0 = wmma_bf16(h[kb], b0, acc0);
      acc1 = wmma_bf16(h[kb], b1, acc1);
      acc2 = wmma_bf16(h[kb], b2, acc2);
      acc3 = wmma_bf16(h[kb], b3, acc3);
    }
    {
      v8f acc[4] = {acc0, acc1, acc2, acc3};
#pragma unroll
      for (int nb = 0; nb < 4; ++nb) {
        int n = nb * 16 + nl;
        float bias = msg_b2[n];
#pragma unroll
        for (int r = 0; r < 8; ++r) {
          int mr = r + mrb;
          float v = (acc[nb][r] + bias) * s_fw[wave][mr];
          atomicAdd(&agg[(long)s_col[wave][mr] * HD + n], v);
        }
      }
    }
  }
}

// ---------------------------------------------------------------------------
// Update kernel (persistent): one wave = one 16-node tile per iteration.
//   cat2 = [x , agg/clip(norm)]   (2 bf16 frags + 2 cvt-from-fp32 frags)
//   h    = relu(cat2 @ upd_W1 + b1)
//   x    = relu(h @ upd_W2 + b2 + x)   rewrite fp32 + bf16 in place
// ---------------------------------------------------------------------------
__global__ void __launch_bounds__(128)
update_kernel(float* __restrict__ xf, unsigned short* __restrict__ xb,
              const float* __restrict__ agg, const float* __restrict__ nrm,
              const unsigned short* __restrict__ pw_upd1, const float* __restrict__ upd_b1,
              const unsigned short* __restrict__ pw_upd2, const float* __restrict__ upd_b2,
              int N, int ntiles) {
  __shared__ unsigned short w_upd1[PM1];   // 16 KB
  __shared__ unsigned short w_upd2[PM2];   //  8 KB
  __shared__ unsigned short s_h1[WPB][16 * HD];

  const int tid = threadIdx.x;
  const int wave = tid >> 5;
  const int lane = tid & 31;

  __builtin_prefetch(pw_upd1, 0, 2);
  __builtin_prefetch(pw_upd2, 0, 2);
  stage_lds(w_upd1, pw_upd1, PM1, tid, 128);
  stage_lds(w_upd2, pw_upd2, PM2, tid, 128);
  __syncthreads();

  const int m = lane & 15;
  const int mrb = (lane >> 4) << 3;
  const int nl = lane & 15;

  for (int tile = blockIdx.x * WPB + wave; tile < ntiles;
       tile += gridDim.x * WPB) {
    long node = (long)tile * 16 + m;
    if (node >= N) node = (long)N - 1;   // clamp for EXEC-uniform WMMA

    const unsigned short* px = xb + node * HD;
    v16bf a[4];
    a[0] = load_A_bf16(px, 0, lane);
    a[1] = load_A_bf16(px, 32, lane);
    float inv = 1.f / fmaxf(nrm[node], 1e-6f);
    const float* pa = agg + node * HD;
    a[2] = load_A_f32_scaled(pa, 0, lane, inv);
    a[3] = load_A_f32_scaled(pa, 32, lane, inv);

    v8f accM0 = {}, accM1 = {}, accM2 = {}, accM3 = {};
#pragma unroll
    for (int kb = 0; kb < 4; ++kb) {
      v16bf b0 = load_B_frag(w_upd1, kb * 4 + 0, lane);
      v16bf b1 = load_B_frag(w_upd1, kb * 4 + 1, lane);
      v16bf b2 = load_B_frag(w_upd1, kb * 4 + 2, lane);
      v16bf b3 = load_B_frag(w_upd1, kb * 4 + 3, lane);
      accM0 = wmma_bf16(a[kb], b0, accM0);
      accM1 = wmma_bf16(a[kb], b1, accM1);
      accM2 = wmma_bf16(a[kb], b2, accM2);
      accM3 = wmma_bf16(a[kb], b3, accM3);
    }
    {
      v8f accM[4] = {accM0, accM1, accM2, accM3};
#pragma unroll
      for (int nb = 0; nb < 4; ++nb) {
        int n = nb * 16 + nl;
        float bias = upd_b1[n];
#pragma unroll
        for (int r = 0; r < 8; ++r)
          s_h1[wave][(r + mrb) * HD + n] = f2bf(fmaxf(accM[nb][r] + bias, 0.f));
      }
    }
    asm volatile("s_wait_dscnt 0" ::: "memory");

    const unsigned short* hrow = &s_h1[wave][m * HD];
    v16bf h[2];
    h[0] = load_A_bf16(hrow, 0, lane);
    h[1] = load_A_bf16(hrow, 32, lane);
    v8f acc0 = {}, acc1 = {}, acc2 = {}, acc3 = {};
#pragma unroll
    for (int kb = 0; kb < 2; ++kb) {
      v16bf b0 = load_B_frag(w_upd2, kb * 4 + 0, lane);
      v16bf b1 = load_B_frag(w_upd2, kb * 4 + 1, lane);
      v16bf b2 = load_B_frag(w_upd2, kb * 4 + 2, lane);
      v16bf b3 = load_B_frag(w_upd2, kb * 4 + 3, lane);
      acc0 = wmma_bf16(h[kb], b0, acc0);
      acc1 = wmma_bf16(h[kb], b1, acc1);
      acc2 = wmma_bf16(h[kb], b2, acc2);
      acc3 = wmma_bf16(h[kb], b3, acc3);
    }
    {
      v8f acc[4] = {acc0, acc1, acc2, acc3};
#pragma unroll
      for (int nb = 0; nb < 4; ++nb) {
        int n = nb * 16 + nl;
        float bias = upd_b2[n];
#pragma unroll
        for (int r = 0; r < 8; ++r) {
          long nd = (long)tile * 16 + (r + mrb);
          if (nd < N) {
            long idx = nd * HD + n;
            float v = fmaxf(acc[nb][r] + bias + xf[idx], 0.f);  // residual+relu
            xf[idx] = v;
            xb[idx] = f2bf(v);
          }
        }
      }
    }
  }
}

// Column-wise max over nodes (x >= 0 after relu -> uint bit-compare == float).
__global__ void max_kernel(const float* __restrict__ xf,
                           unsigned* __restrict__ smax, int N) {
  __shared__ float red[256];
  int h = threadIdx.x & 63;
  int sub = threadIdx.x >> 6;
  long start = (long)blockIdx.x * 1024;
  long end = start + 1024; if (end > N) end = N;
  float mx = 0.f;
  for (long rr = start + sub; rr < end; rr += 4)
    mx = fmaxf(mx, xf[rr * HD + h]);
  red[threadIdx.x] = mx;
  __syncthreads();
  if (sub == 0) {
    mx = fmaxf(fmaxf(red[h], red[64 + h]), fmaxf(red[128 + h], red[192 + h]));
    atomicMax(&smax[h], __float_as_uint(mx));
  }
}

__global__ void head_kernel(const unsigned* __restrict__ smax,
                            const float* __restrict__ hW,
                            const float* __restrict__ hb,
                            float* __restrict__ out) {
  __shared__ float red[64];
  int h = threadIdx.x;
  red[h] = __uint_as_float(smax[h]) * hW[h];
  __syncthreads();
  for (int s = 32; s > 0; s >>= 1) {
    if (h < s) red[h] += red[h + s];
    __syncthreads();
  }
  if (h == 0) out[0] = red[0] + hb[0];
}

// ---------------------------------------------------------------------------
extern "C" void kernel_launch(void* const* d_in, const int* in_sizes, int n_in,
                              void* d_out, int out_size, void* d_ws, size_t ws_size,
                              hipStream_t stream) {
  const float* node_features = (const float*)d_in[0];
  const int*   edge_index    = (const int*)d_in[1];
  const float* edge_weights  = (const float*)d_in[2];
  const float* embed_W = (const float*)d_in[3];
  const float* embed_b = (const float*)d_in[4];
  const float* msg_W1 = (const float*)d_in[5];
  const float* msg_b1 = (const float*)d_in[6];
  const float* msg_W2 = (const float*)d_in[7];
  const float* msg_b2 = (const float*)d_in[8];
  const float* att_W1 = (const float*)d_in[9];
  const float* att_b1 = (const float*)d_in[10];
  const float* att_W2 = (const float*)d_in[11];
  const float* att_b2 = (const float*)d_in[12];
  const float* upd_W1 = (const float*)d_in[13];
  const float* upd_b1 = (const float*)d_in[14];
  const float* upd_W2 = (const float*)d_in[15];
  const float* upd_b2 = (const float*)d_in[16];
  const float* head_W = (const float*)d_in[17];
  const float* head_b = (const float*)d_in[18];

  const int N = in_sizes[0] / NF;
  const int E = in_sizes[2];
  const int* row = edge_index;
  const int* col = edge_index + E;

  // workspace carving
  char* ws = (char*)d_ws;
  size_t off = 0;
  auto carve = [&](size_t bytes) -> void* {
    void* p = ws + off;
    off = (off + bytes + 255) & ~(size_t)255;
    return p;
  };
  float*          xf   = (float*)carve((size_t)N * HD * sizeof(float));
  unsigned short* xb   = (unsigned short*)carve((size_t)N * HD * sizeof(unsigned short));
  float*          agg  = (float*)carve((size_t)N * HD * sizeof(float));
  float*          nrm  = (float*)carve((size_t)N * sizeof(float));
  unsigned*       smax = (unsigned*)carve(HD * sizeof(unsigned));
  unsigned short* pw_msg1[2]; unsigned short* pw_msg2[2];
  unsigned short* pw_att1[2]; unsigned short* pw_upd1[2]; unsigned short* pw_upd2[2];
  for (int l = 0; l < 2; ++l) {
    pw_msg1[l] = (unsigned short*)carve(PM1 * 2);
    pw_msg2[l] = (unsigned short*)carve(PM2 * 2);
    pw_att1[l] = (unsigned short*)carve(PA1 * 2);
    pw_upd1[l] = (unsigned short*)carve(PM1 * 2);
    pw_upd2[l] = (unsigned short*)carve(PM2 * 2);
  }

  // 1) pack weights
  for (int l = 0; l < 2; ++l) {
    pack_weights_kernel<<<(PM1 + 255) / 256, 256, 0, stream>>>(msg_W1 + (size_t)l * 128 * 64, pw_msg1[l], 128, 64);
    pack_weights_kernel<<<(PM2 + 255) / 256, 256, 0, stream>>>(msg_W2 + (size_t)l * 64 * 64,  pw_msg2[l], 64, 64);
    pack_weights_kernel<<<(PA1 + 255) / 256, 256, 0, stream>>>(att_W1 + (size_t)l * 128 * 32, pw_att1[l], 128, 32);
    pack_weights_kernel<<<(PM1 + 255) / 256, 256, 0, stream>>>(upd_W1 + (size_t)l * 128 * 64, pw_upd1[l], 128, 64);
    pack_weights_kernel<<<(PM2 + 255) / 256, 256, 0, stream>>>(upd_W2 + (size_t)l * 64 * 64,  pw_upd2[l], 64, 64);
  }

  // 2) embedding
  {
    long total = (long)N * HD;
    embed_kernel<<<(unsigned)((total + 255) / 256), 256, 0, stream>>>(
        node_features, embed_W, embed_b, xf, xb, N);
  }

  // 3) message-passing layers (persistent grids)
  const int etiles = (E + 15) / 16;
  const int eblocks_full = (etiles + WPB - 1) / WPB;
  const int eblocks = eblocks_full < 2048 ? eblocks_full : 2048;
  const int ntiles_n = (N + 15) / 16;
  const int nblocks_full = (ntiles_n + WPB - 1) / WPB;
  const int nblocks = nblocks_full < 2048 ? nblocks_full : 2048;
  for (int l = 0; l < 2; ++l) {
    long zlen = (long)N * HD;
    zero_f32_kernel<<<(unsigned)((zlen + 255) / 256), 256, 0, stream>>>(agg, zlen);
    zero_f32_kernel<<<(unsigned)((N + 255) / 256), 256, 0, stream>>>(nrm, N);
    edge_kernel<<<eblocks, 128, 0, stream>>>(
        xb, row, col, edge_weights,
        pw_msg1[l], msg_b1 + (size_t)l * HD,
        pw_msg2[l], msg_b2 + (size_t)l * HD,
        pw_att1[l], att_b1 + (size_t)l * 32,
        att_W2 + (size_t)l * 32, att_b2 + (size_t)l,
        agg, nrm, E, etiles);
    update_kernel<<<nblocks, 128, 0, stream>>>(
        xf, xb, agg, nrm,
        pw_upd1[l], upd_b1 + (size_t)l * HD,
        pw_upd2[l], upd_b2 + (size_t)l * HD,
        N, ntiles_n);
  }

  // 4) global max-pool + head
  zero_u32_kernel<<<1, 64, 0, stream>>>(smax, HD);
  max_kernel<<<(unsigned)((N + 1023) / 1024), 256, 0, stream>>>(xf, smax, N);
  head_kernel<<<1, 64, 0, stream>>>(smax, head_W, head_b, (float*)d_out);
}
